// HopfieldAttention_16535624090005
// MI455X (gfx1250) — matrix operand order repaired and checked by reading, not compiled, not run on hardware
//
#include <hip/hip_runtime.h>
#include <hip/hip_bf16.h>

#define B_   2
#define S_   2048
#define HID_ 1024
#define NH_  16
#define HD_  64

typedef __bf16 bf16;
typedef __attribute__((ext_vector_type(16))) __bf16 v16bf;
typedef __attribute__((ext_vector_type(8)))  __bf16 v8bf;
typedef __attribute__((ext_vector_type(8)))  float  v8f;

static __device__ __forceinline__ v8f wmma_bf16(v16bf a, v16bf b, v8f c) {
  // (neg_a, A, neg_b, B, c_mod, C, reuse_a, reuse_b)
  return __builtin_amdgcn_wmma_f32_16x16x32_bf16(false, a, false, b, (short)0, c,
                                                 false, false);
}

// ---------------------------------------------------------------- convert
__global__ void f32_to_bf16(const float* __restrict__ src, bf16* __restrict__ dst,
                            int n) {
  int i = blockIdx.x * blockDim.x + threadIdx.x;
  const int stride = gridDim.x * blockDim.x;
  for (; i < n; i += stride) dst[i] = (bf16)src[i];
}

// ---------------------------------------------------------------- projection GEMM
// C[4096 x 1024] = X[4096 x 1024(bf16)] * W^T (W row-major [n][k], bf16) + bias
// Each wave computes a 16(M) x 64(N) strip: one shared A fragment, 4 accumulators,
// software-pipelined K loop (next fragments load while current WMMAs execute).
// mode 0: Q   -> outA[b][h][s][d]  (scaled by 1/sqrt(HD))
// mode 1: K   -> outA[b][h][s][d]  and outB = K^T [b][h][d][s]   (mask folded)
// mode 2: V   -> outB = V^T [b][h][d][s]                         (mask folded)
// mode 3: out -> outF row-major f32 (B,S,HID)
__global__ __launch_bounds__(256)
void gemm_proj(const bf16* __restrict__ X, const bf16* __restrict__ W,
               const float* __restrict__ bias, const float* __restrict__ mask,
               bf16* __restrict__ outA, bf16* __restrict__ outB,
               float* __restrict__ outF, int mode, float scale) {
  const int lane = threadIdx.x & 31;
  const int wv   = threadIdx.x >> 5;
  const int g    = blockIdx.x * 8 + wv;       // one 16x64 strip per wave
  const int mt   = g >> 4;                    // 256 row tiles
  const int nt   = g & 15;                    // 16 column strips (64 wide)
  const int half = lane >> 4;
  const int col  = lane & 15;

  const bf16*  ap = X + (size_t)(mt * 16 + col) * HID_ + half * 16;
  const bf16*  bp = W + (size_t)(nt * 64 + col) * HID_ + half * 16;
  const size_t BJ = (size_t)16 * HID_;        // 16 W-rows between N-tiles

#define LOAD_A(dst, k)                                      \
  {                                                         \
    ((v8bf*)&(dst))[0] = *(const v8bf*)(ap + (k));          \
    ((v8bf*)&(dst))[1] = *(const v8bf*)(ap + (k) + 8);      \
  }

  v16bf a, b0, b1, b2, b3;
  LOAD_A(a, 0);
  b0 = *(const v16bf*)(bp);
  b1 = *(const v16bf*)(bp + BJ);
  b2 = *(const v16bf*)(bp + 2 * BJ);
  b3 = *(const v16bf*)(bp + 3 * BJ);

  v8f acc0 = {}, acc1 = {}, acc2 = {}, acc3 = {};
#pragma unroll 2
  for (int kb = 32; kb < HID_; kb += 32) {
    v16bf an, c0, c1, c2, c3;
    LOAD_A(an, kb);
    c0 = *(const v16bf*)(bp + kb);
    c1 = *(const v16bf*)(bp + BJ + kb);
    c2 = *(const v16bf*)(bp + 2 * BJ + kb);
    c3 = *(const v16bf*)(bp + 3 * BJ + kb);
    __builtin_prefetch(bp + kb + 480, 0, 1);  // speculative; OOB is dropped
    acc0 = wmma_bf16(a, b0, acc0);
    acc1 = wmma_bf16(a, b1, acc1);
    acc2 = wmma_bf16(a, b2, acc2);
    acc3 = wmma_bf16(a, b3, acc3);
    a = an; b0 = c0; b1 = c1; b2 = c2; b3 = c3;
  }
  acc0 = wmma_bf16(a, b0, acc0);
  acc1 = wmma_bf16(a, b1, acc1);
  acc2 = wmma_bf16(a, b2, acc2);
  acc3 = wmma_bf16(a, b3, acc3);
#undef LOAD_A

  const int h = nt;                            // 64-wide strip == one head
  v8f accs[4] = {acc0, acc1, acc2, acc3};
#pragma unroll
  for (int jt = 0; jt < 4; ++jt) {
    const int   ncol = nt * 64 + jt * 16 + col;
    const int   d    = ncol & 63;
    const float bval = bias[ncol];
#pragma unroll
    for (int r = 0; r < 8; ++r) {
      const int row = mt * 16 + half * 8 + r;  // 0..4095
      const int bb  = row >> 11;               // row / S
      const int sr  = row & (S_ - 1);
      float v = (accs[jt][r] + bval) * scale;
      if (mode == 1 || mode == 2) v *= mask[row];
      const size_t bhs = ((size_t)(bb * NH_ + h) * S_ + sr);
      const size_t bhd = ((size_t)(bb * NH_ + h) * HD_ + d);
      if (mode == 0) {
        outA[(bhs << 6) + d] = (bf16)v;
      } else if (mode == 1) {
        outA[(bhs << 6) + d] = (bf16)v;
        outB[bhd * S_ + sr]  = (bf16)v;
      } else if (mode == 2) {
        outB[bhd * S_ + sr]  = (bf16)v;
      } else {
        outF[(size_t)row * HID_ + ncol] = v;
      }
    }
  }
}

// ---------------------------------------------------------------- flash attention
// One wave per 16-row Q tile; streams 32-key blocks with online softmax.
// Q:[bh][S][HD] (pre-scaled), K:[bh][S][HD], Vt:[bh][HD][S]; O:(B,S,HID) bf16.
__global__ __launch_bounds__(128)
void flash_fwd(const bf16* __restrict__ Q, const bf16* __restrict__ K,
               const bf16* __restrict__ Vt, const float* __restrict__ amask,
               bf16* __restrict__ O) {
  __shared__ bf16 Pb[4][16 * 32];
  const int lane = threadIdx.x & 31;
  const int wv   = threadIdx.x >> 5;
  const int g    = blockIdx.x * 4 + wv;       // 0..4095
  const int t    = g & 127;                   // q tile index
  const int bh   = g >> 7;
  const int b    = bh >> 4;
  const int h    = bh & 15;
  const int q0   = t * 16;
  const int half = lane >> 4;
  const int col  = lane & 15;

  const bf16* Qb = Q  + (size_t)bh * S_ * HD_;
  const bf16* Kb = K  + (size_t)bh * S_ * HD_;
  const bf16* Vb = Vt + (size_t)bh * HD_ * S_;

  v16bf A0, A1;                               // Q fragments, d-chunks 0..31 / 32..63
  {
    const bf16* qp = Qb + (size_t)(q0 + col) * HD_ + half * 16;
    ((v8bf*)&A0)[0] = *(const v8bf*)(qp);
    ((v8bf*)&A0)[1] = *(const v8bf*)(qp + 8);
    ((v8bf*)&A1)[0] = *(const v8bf*)(qp + 32);
    ((v8bf*)&A1)[1] = *(const v8bf*)(qp + 40);
  }

  v8f O0 = {}, O1 = {}, O2 = {}, O3 = {};
  float mrow[8], lrow[8];
#pragma unroll
  for (int r = 0; r < 8; ++r) { mrow[r] = -__builtin_inff(); lrow[r] = 0.f; }

  const int qbase = q0 + half * 8;

  for (int jb = 0; jb < q0 + 16; jb += 32) {
    // ---- scores: two 16x16 tiles (keys jb..jb+15 and jb+16..jb+31)
    v8f s0 = {}, s1 = {};
    const bf16* kp0 = Kb + (size_t)(jb + col) * HD_ + half * 16;
    const bf16* kp1 = kp0 + 16 * HD_;
    v16bf B00 = *(const v16bf*)(kp0);
    v16bf B01 = *(const v16bf*)(kp0 + 32);
    v16bf B10 = *(const v16bf*)(kp1);
    v16bf B11 = *(const v16bf*)(kp1 + 32);
    s0 = wmma_bf16(A0, B00, s0);
    s0 = wmma_bf16(A1, B01, s0);
    s1 = wmma_bf16(A0, B10, s1);
    s1 = wmma_bf16(A1, B11, s1);

    const int  kc0 = jb + col;
    const int  kc1 = kc0 + 16;
    const bool z0  = (amask[b * S_ + kc0] == 0.f);
    const bool z1  = (amask[b * S_ + kc1] == 0.f);

    bf16* prow = &Pb[wv][half * 8 * 32];
#pragma unroll
    for (int r = 0; r < 8; ++r) {
      const int qr = qbase + r;
      float v0 = s0[r];
      float v1 = s1[r];
      if (z0 || kc0 > qr) v0 = -__builtin_inff();   // causal + padding mask
      if (z1 || kc1 > qr) v1 = -__builtin_inff();
      float mx = fmaxf(v0, v1);
      mx = fmaxf(mx, __shfl_xor(mx, 1, 32));
      mx = fmaxf(mx, __shfl_xor(mx, 2, 32));
      mx = fmaxf(mx, __shfl_xor(mx, 4, 32));
      mx = fmaxf(mx, __shfl_xor(mx, 8, 32));
      const float mn    = fmaxf(mrow[r], mx);
      const float alpha = __expf(mrow[r] - mn);
      const float p0    = __expf(v0 - mn);
      const float p1    = __expf(v1 - mn);
      float ps = p0 + p1;
      ps += __shfl_xor(ps, 1, 32);
      ps += __shfl_xor(ps, 2, 32);
      ps += __shfl_xor(ps, 4, 32);
      ps += __shfl_xor(ps, 8, 32);
      lrow[r] = lrow[r] * alpha + ps;
      mrow[r] = mn;
      O0[r] *= alpha; O1[r] *= alpha; O2[r] *= alpha; O3[r] *= alpha;
      prow[r * 32 + col]      = (bf16)p0;           // P tile, row-major 16x32
      prow[r * 32 + col + 16] = (bf16)p1;
    }
    // cross-lane LDS RAW: wait for stores, then reload in A-fragment layout
    asm volatile("s_wait_dscnt 0" ::: "memory");

    v16bf PA;
    const bf16* pl = &Pb[wv][col * 32 + half * 16];
    ((v8bf*)&PA)[0] = *(const v8bf*)(pl);
    ((v8bf*)&PA)[1] = *(const v8bf*)(pl + 8);

    // ---- O += P(16x32) * V(32x64), B fragments from V^T rows (contiguous)
    const bf16* vp = Vb + (size_t)col * S_ + jb + half * 16;
    v16bf V0 = *(const v16bf*)(vp);
    v16bf V1 = *(const v16bf*)(vp + 16 * S_);
    v16bf V2 = *(const v16bf*)(vp + 32 * S_);
    v16bf V3 = *(const v16bf*)(vp + 48 * S_);
    O0 = wmma_bf16(PA, V0, O0);
    O1 = wmma_bf16(PA, V1, O1);
    O2 = wmma_bf16(PA, V2, O2);
    O3 = wmma_bf16(PA, V3, O3);
  }

#pragma unroll
  for (int r = 0; r < 8; ++r) {
    const float  inv = 1.0f / lrow[r];
    const size_t row = (size_t)b * S_ + qbase + r;
    bf16* op = O + row * HID_ + h * 64 + col;
    op[0]  = (bf16)(O0[r] * inv);
    op[16] = (bf16)(O1[r] * inv);
    op[32] = (bf16)(O2[r] * inv);
    op[48] = (bf16)(O3[r] * inv);
  }
}

// ---------------------------------------------------------------- Hopfield state
// state[bh][d][e] = sum_s K[s][d] * V[s][e]; one wave per 16x16 tile of the 64x64.
__global__ __launch_bounds__(256)
void state_k(const bf16* __restrict__ Kt, const bf16* __restrict__ Vt,
             float* __restrict__ out) {
  const int lane = threadIdx.x & 31;
  const int wv   = threadIdx.x >> 5;
  const int g    = blockIdx.x * 8 + wv;       // 512 tiles total
  const int tile = g & 15;
  const int bh   = g >> 4;
  const int dt   = tile >> 2;
  const int et   = tile & 3;
  const int half = lane >> 4;
  const int col  = lane & 15;

  const bf16* kp = Kt + ((size_t)bh * HD_ + dt * 16 + col) * S_ + half * 16;
  const bf16* vp = Vt + ((size_t)bh * HD_ + et * 16 + col) * S_ + half * 16;

  v8f acc = {};
#pragma unroll 2
  for (int s = 0; s < S_; s += 32) {
    v16bf a, b;
    ((v8bf*)&a)[0] = *(const v8bf*)(kp + s);
    ((v8bf*)&a)[1] = *(const v8bf*)(kp + s + 8);
    b = *(const v16bf*)(vp + s);
    acc = wmma_bf16(a, b, acc);
  }
#pragma unroll
  for (int r = 0; r < 8; ++r) {
    out[((size_t)bh * HD_ + dt * 16 + half * 8 + r) * HD_ + et * 16 + col] = acc[r];
  }
}

// ---------------------------------------------------------------- launcher
extern "C" void kernel_launch(void* const* d_in, const int* in_sizes, int n_in,
                              void* d_out, int out_size, void* d_ws, size_t ws_size,
                              hipStream_t stream) {
  (void)in_sizes; (void)n_in; (void)out_size; (void)ws_size;
  const float* hs    = (const float*)d_in[0];
  const float* amask = (const float*)d_in[1];
  const float* Wq    = (const float*)d_in[2];
  const float* bq    = (const float*)d_in[3];
  const float* Wk    = (const float*)d_in[4];
  const float* bk    = (const float*)d_in[5];
  const float* Wv    = (const float*)d_in[6];
  const float* bv    = (const float*)d_in[7];
  const float* Wo    = (const float*)d_in[8];
  const float* bo    = (const float*)d_in[9];

  const size_t NX = (size_t)B_ * S_ * HID_;   // 4,194,304
  const size_t NW = (size_t)HID_ * HID_;      // 1,048,576

  bf16* ws    = (bf16*)d_ws;
  bf16* xb    = ws;
  bf16* wqb   = xb + NX;
  bf16* wkb   = wqb + NW;
  bf16* wvb   = wkb + NW;
  bf16* wob   = wvb + NW;
  bf16* q_ws  = wob + NW;
  bf16* k_ws  = q_ws + NX;
  bf16* kt_ws = k_ws + NX;
  bf16* vt_ws = kt_ws + NX;
  bf16* o_ws  = vt_ws + NX;

  float* outF   = (float*)d_out;
  float* stateF = outF + NX;

  f32_to_bf16<<<1024, 256, 0, stream>>>(hs, xb, (int)NX);
  f32_to_bf16<<<512, 256, 0, stream>>>(Wq, wqb, (int)NW);
  f32_to_bf16<<<512, 256, 0, stream>>>(Wk, wkb, (int)NW);
  f32_to_bf16<<<512, 256, 0, stream>>>(Wv, wvb, (int)NW);
  f32_to_bf16<<<512, 256, 0, stream>>>(Wo, wob, (int)NW);

  const int gemmBlocks = (4096 / 16) * (HID_ / 64) / 8;  // 512
  const float qscale = 0.125f;                           // 1/sqrt(HD), TEMP=1
  gemm_proj<<<gemmBlocks, 256, 0, stream>>>(xb, wqb, bq, nullptr, q_ws, nullptr,
                                            nullptr, 0, qscale);
  gemm_proj<<<gemmBlocks, 256, 0, stream>>>(xb, wkb, bk, amask, k_ws, kt_ws,
                                            nullptr, 1, 1.0f);
  gemm_proj<<<gemmBlocks, 256, 0, stream>>>(xb, wvb, bv, amask, nullptr, vt_ws,
                                            nullptr, 2, 1.0f);

  flash_fwd<<<(B_ * NH_ * (S_ / 16)) / 4, 128, 0, stream>>>(q_ws, k_ws, vt_ws,
                                                            amask, o_ws);
  state_k<<<B_ * NH_ * 16 / 8, 256, 0, stream>>>(kt_ws, vt_ws, stateF);

  gemm_proj<<<gemmBlocks, 256, 0, stream>>>(o_ws, wob, bo, nullptr, nullptr,
                                            nullptr, outF, 3, 1.0f);
}
